// MambaVisionMixer_22187801051651
// MI455X (gfx1250) — matrix operand order repaired and checked by reading, not compiled, not run on hardware
//
#include <hip/hip_runtime.h>
#include <hip/hip_bf16.h>
#include <math.h>

// ---------------- problem constants (from reference) ----------------
#define D_MODEL  512
#define D_INNER  1024
#define D_HALF   512
#define DT_RANK  32
#define D_STATE  16
#define D_CONV   4
#define B_SZ     8
#define L_SEQ    4096
#define M_ROWS   (B_SZ * L_SEQ)   // 32768 flattened (b,l) rows

// ---------------- CDNA5 WMMA / TDM types ----------------
typedef __bf16 bf16;
typedef __bf16 bf16x16 __attribute__((ext_vector_type(16)));
typedef float  f32x8   __attribute__((ext_vector_type(8)));
typedef unsigned int u32x4v __attribute__((ext_vector_type(4)));
typedef int          i32x4v __attribute__((ext_vector_type(4)));
typedef int          i32x8v __attribute__((ext_vector_type(8)));

union Frag { uint4 q[2]; bf16x16 v; unsigned short u[16]; };

__device__ __forceinline__ unsigned short f32_to_bf16_rne(float f) {
  unsigned int u = __float_as_uint(f);
  unsigned int r = 0x7FFFu + ((u >> 16) & 1u);
  return (unsigned short)((u + r) >> 16);
}

// =====================================================================
// Pack f32 weights (K x N row-major) into bf16 WMMA B-fragment order.
// fb = kb*(N/16) + nt; lane l: col = nt*16 + (l&15), K = kb*32+(l>>4)*8+i (+8 if i>=8)
// Storage: out[(fb*32 + lane)*16 + i]  -> one 32B load/lane in the GEMM.
// =====================================================================
__global__ void mamba_pack_w_bf16(const float* __restrict__ W, int K, int N,
                                  unsigned short* __restrict__ out) {
  int id = blockIdx.x * blockDim.x + threadIdx.x;
  int total = (K >> 5) * (N >> 4) * 32;
  if (id >= total) return;
  int lane   = id & 31;
  int ft     = id >> 5;
  int nTiles = N >> 4;
  int nt = ft % nTiles;
  int kb = ft / nTiles;
  int k0  = kb * 32 + ((lane >> 4) << 3);
  int col = (nt << 4) + (lane & 15);
  unsigned short* dst = out + (size_t)id * 16;
#pragma unroll
  for (int i = 0; i < 16; ++i)
    dst[i] = f32_to_bf16_rne(W[(size_t)(k0 + i + ((i >> 3) << 3)) * N + col]);
}

// =====================================================================
// Pack f32 activations (M x K, leading dim lda) into bf16 WMMA A-fragment
// order (ISA A 16x32 bf16 layout). fa = rowblk*(K/32) + kb; lane l:
//   row = rowblk*16 + (l&15); i<8 -> K=kb*32+(l>>4)*8+i; i>=8 -> +16.
// =====================================================================
__global__ void mamba_pack_a_bf16(const float* __restrict__ A, int lda, int K,
                                  unsigned short* __restrict__ out, int total) {
  int id = blockIdx.x * blockDim.x + threadIdx.x;
  if (id >= total) return;
  int lane   = id & 31;
  int fa     = id >> 5;
  int kTiles = K >> 5;
  int kb     = fa % kTiles;
  int rowblk = fa / kTiles;
  int row    = rowblk * 16 + (lane & 15);
  int kbase  = kb * 32 + ((lane >> 4) << 3);
  const float* ap = A + (size_t)row * lda + kbase;
  float4 a0 = *(const float4*)(ap + 0);
  float4 a1 = *(const float4*)(ap + 4);
  float4 a2 = *(const float4*)(ap + 16);
  float4 a3 = *(const float4*)(ap + 20);
  unsigned short* dst = out + (size_t)id * 16;
  dst[0]  = f32_to_bf16_rne(a0.x); dst[1]  = f32_to_bf16_rne(a0.y);
  dst[2]  = f32_to_bf16_rne(a0.z); dst[3]  = f32_to_bf16_rne(a0.w);
  dst[4]  = f32_to_bf16_rne(a1.x); dst[5]  = f32_to_bf16_rne(a1.y);
  dst[6]  = f32_to_bf16_rne(a1.z); dst[7]  = f32_to_bf16_rne(a1.w);
  dst[8]  = f32_to_bf16_rne(a2.x); dst[9]  = f32_to_bf16_rne(a2.y);
  dst[10] = f32_to_bf16_rne(a2.z); dst[11] = f32_to_bf16_rne(a2.w);
  dst[12] = f32_to_bf16_rne(a3.x); dst[13] = f32_to_bf16_rne(a3.y);
  dst[14] = f32_to_bf16_rne(a3.z); dst[15] = f32_to_bf16_rne(a3.w);
}

// =====================================================================
// WMMA GEMM on pre-packed bf16 fragments:
//   C[M x N](f32) = Ap[M x K] * Bp[K x N]
// Block = 256 threads = 8 waves stacked in M; wave tile = 16(M) x 64(N).
// Inner loop is pure b128 loads + v_wmma (weights are L2-resident).
// EPI: 0 = none, 1 = softplus(acc + bias[col])
// =====================================================================
template <int EPI>
__global__ __launch_bounds__(256) void mamba_gemm_wmma_bf16(
    const unsigned short* __restrict__ Ap,
    const unsigned short* __restrict__ Bp,
    float* __restrict__ C, int ldc, int K, int N,
    const float* __restrict__ bias) {
  const int lane     = threadIdx.x & 31;
  const int wave     = threadIdx.x >> 5;
  const int row_base = blockIdx.y * 128 + wave * 16;
  const int col_base = blockIdx.x * 64;
  const int nTiles   = N >> 4;
  const int kTiles   = K >> 5;
  const int m_lane   = lane & 15;
  const int hi       = lane >> 4;

  f32x8 acc[4] = {};

  // packed-A base for this wave (fragments contiguous along kb)
  const size_t aBase = ((size_t)(blockIdx.y * 8 + wave) * kTiles * 32 + lane) * 16;

  for (int kb = 0; kb < kTiles; ++kb) {
    Frag af;
    const uint4* aq = (const uint4*)(Ap + aBase + (size_t)kb * 512);
    af.q[0] = aq[0];
    af.q[1] = aq[1];

    size_t fb = (size_t)kb * nTiles + (col_base >> 4);
    const uint4* bp = (const uint4*)(Bp + (fb * 32 + lane) * 16);
#pragma unroll
    for (int t = 0; t < 4; ++t) {
      Frag bf;
      bf.q[0] = bp[t * 64 + 0];     // tile stride = 32 lanes * 32B = 64 uint4
      bf.q[1] = bp[t * 64 + 1];
      acc[t] = __builtin_amdgcn_wmma_f32_16x16x32_bf16(
          false, af.v, false, bf.v, (short)0, acc[t], false, false);
    }
  }

  // ---- epilogue + store (D layout: VGPR r -> rows r / r+8) ----
#pragma unroll
  for (int t = 0; t < 4; ++t) {
    int col = col_base + t * 16 + m_lane;
    float bv = (EPI == 1) ? bias[col] : 0.f;
#pragma unroll
    for (int r = 0; r < 8; ++r) {
      int row = row_base + hi * 8 + r;
      float v = acc[t][r];
      if (EPI == 1) {                      // softplus(v + b_dt)
        v += bv;
        v = (v > 20.f) ? v : log1pf(__expf(v));
      }
      C[(size_t)row * ldc + col] = v;
    }
  }
}

// =====================================================================
// Depthwise conv (k=4, SAME: pad_lo=1) + SiLU for both x and z halves.
// =====================================================================
__global__ __launch_bounds__(256) void mamba_dwconv_silu(
    const float* __restrict__ xz, const float* __restrict__ Kx,
    const float* __restrict__ Kz, float* __restrict__ x_conv,
    float* __restrict__ yz) {
  size_t idx = (size_t)blockIdx.x * blockDim.x + threadIdx.x;
  if (idx >= (size_t)M_ROWS * D_HALF) return;
  int d = (int)(idx % D_HALF);
  size_t rowflat = idx / D_HALF;
  int t = (int)(rowflat % L_SEQ);
  size_t bbase = (rowflat / L_SEQ) * (size_t)L_SEQ;

  float ax = 0.f, az = 0.f;
#pragma unroll
  for (int j = 0; j < D_CONV; ++j) {
    int tt = t - 1 + j;                 // TF 'SAME', k=4 -> pad_lo = 1
    if (tt >= 0 && tt < L_SEQ) {
      size_t rr = (bbase + (size_t)tt) * D_INNER;
      ax += xz[rr + d]          * Kx[d * 4 + j];
      az += xz[rr + D_HALF + d] * Kz[d * 4 + j];
    }
  }
  float sx = ax / (1.f + __expf(-ax));
  float sz = az / (1.f + __expf(-az));
  x_conv[rowflat * D_HALF + d] = sx;
  yz[rowflat * (size_t)D_INNER + D_HALF + d] = sz;
}

// =====================================================================
// Selective scan. 1 block per batch (512 threads = 1 lane per channel d).
// B/C chunk (32x64 f32 tile, row stride 64) staged global->LDS via the
// Tensor Data Mover (D# built per ISA 08_async_tensor §8.3/8.4), issued
// by wave 0, completion via s_wait_tensorcnt + workgroup barrier.
// =====================================================================
#define MAMBA_HAVE_TDM __has_builtin(__builtin_amdgcn_tensor_load_to_lds)

__global__ __launch_bounds__(512) void mamba_ssm_scan(
    const float* __restrict__ delta, const float* __restrict__ x_conv,
    const float* __restrict__ x_dbl, const float* __restrict__ A_log,
    const float* __restrict__ Dvec, float* __restrict__ yz) {
  const int b = blockIdx.x;
  const int d = threadIdx.x;

  float A[D_STATE];
#pragma unroll
  for (int n = 0; n < D_STATE; ++n) A[n] = -__expf(A_log[d * D_STATE + n]);
  const float Dd = Dvec[d];

  float h[D_STATE];
#pragma unroll
  for (int n = 0; n < D_STATE; ++n) h[n] = 0.f;

  __shared__ float sBC[64 * 32];        // 64 timesteps x (B[16] ++ C[16])
  const size_t rowb = (size_t)b * L_SEQ;

  for (int t0 = 0; t0 < L_SEQ; t0 += 64) {
#if MAMBA_HAVE_TDM
    if (threadIdx.x < 32) {             // wave 0 drives the TDM
      unsigned long long ga =
          (unsigned long long)(const void*)&x_dbl[(rowb + t0) * 64 + DT_RANK];
      unsigned ldsa = (unsigned)(size_t)(void*)sBC;   // LDS byte offset
      u32x4v g0;
      g0.x = 1u;                                       // count = 1 descriptor
      g0.y = ldsa;                                     // lds_addr
      g0.z = (unsigned)ga;                             // global_addr[31:0]
      g0.w = (unsigned)((ga >> 32) & 0x01FFFFFFu)      // global_addr[56:32]
             | (2u << 30);                             // type = 2 ("image")
      i32x8v g1;
      g1[0] = 0x00020000;          // workgroup_mask=0, data_size=2 (4B)
      g1[1] = (int)(32u << 16);    // tensor_dim0 = 32  (bits 79:48 lo)
      g1[2] = (int)(64u << 16);    // dim0 hi=0 | tensor_dim1 = 64 (lo)
      g1[3] = (int)(32u << 16);    // dim1 hi=0 | tile_dim0 = 32
      g1[4] = 64;                  // tile_dim1 = 64, tile_dim2 = 0
      g1[5] = 64;                  // tensor_dim0_stride = 64 elems (lo 32)
      g1[6] = 0;                   // stride hi | tensor_dim1_stride lo
      g1[7] = 0;
      i32x4v zg4 = {0, 0, 0, 0};   // groups 2/3 unused (<=2D tensor)
      i32x8v zg8 = {0, 0, 0, 0, 0, 0, 0, 0};
      // 6-arg form (clang-23 / therock-10.0 lane, probe-verified arity)
      __builtin_amdgcn_tensor_load_to_lds(g0, g1, zg4, zg4, zg8, 0);
      __builtin_amdgcn_s_wait_tensorcnt(0);
    }
    __syncthreads();
#else
    for (int i = threadIdx.x; i < 64 * 32; i += 512) {
      int tt = t0 + (i >> 5);
      sBC[i] = x_dbl[(rowb + tt) * 64 + DT_RANK + (i & 31)];
    }
    __syncthreads();
#endif
    if (t0 + 64 < L_SEQ)                // gfx1250 global_prefetch of next chunk
      __builtin_prefetch(&x_dbl[(rowb + t0 + 64) * 64], 0, 0);

    for (int tl = 0; tl < 64; ++tl) {
      size_t row = rowb + t0 + tl;
      float dt = delta[row * D_HALF + d];
      float xv = x_conv[row * D_HALF + d];
      float dx = dt * xv;
      const float* Bv = &sBC[tl * 32];
      float y = 0.f;
#pragma unroll
      for (int n = 0; n < D_STATE; ++n) {
        float dA = __expf(dt * A[n]);
        h[n] = h[n] * dA + dx * Bv[n];
        y += h[n] * Bv[16 + n];
      }
      yz[row * D_INNER + d] = y + xv * Dd;
    }
    __syncthreads();
  }
}

// =====================================================================
// Orchestration
// =====================================================================
extern "C" void kernel_launch(void* const* d_in, const int* in_sizes, int n_in,
                              void* d_out, int out_size, void* d_ws,
                              size_t ws_size, hipStream_t stream) {
  (void)in_sizes; (void)n_in; (void)out_size; (void)ws_size;
  const float* hs    = (const float*)d_in[0];
  const float* W_in  = (const float*)d_in[1];
  const float* W_xp  = (const float*)d_in[2];
  const float* W_dt  = (const float*)d_in[3];
  const float* b_dt  = (const float*)d_in[4];
  const float* A_log = (const float*)d_in[5];
  const float* Dvec  = (const float*)d_in[6];
  const float* K_x   = (const float*)d_in[7];
  const float* K_z   = (const float*)d_in[8];
  const float* W_out = (const float*)d_in[9];
  float* out = (float*)d_out;

  // ---- workspace carve-up ----
  float* bufA  = (float*)d_ws;                       // M x 1024 : xz, later x_dbl+delta
  float* bufB  = bufA + (size_t)M_ROWS * D_INNER;    // M x 1024 : [y | z]
  float* xcv   = bufB + (size_t)M_ROWS * D_INNER;    // M x 512  : silu(conv(x))
  float* xdbl  = bufA;                               // M x 64   (after conv)
  float* deltab= bufA + (size_t)M_ROWS * 64;         // M x 512  (after conv)
  unsigned short* pWin  = (unsigned short*)(xcv + (size_t)M_ROWS * D_HALF);
  unsigned short* pWxp  = pWin + (size_t)512 * 1024;
  unsigned short* pWdt  = pWxp + (size_t)512 * 64;
  unsigned short* pWout = pWdt + (size_t)32 * 512;
  unsigned short* pA    = pWout + (size_t)1024 * 512; // M x 1024 bf16, reused serially

  auto packW = [&](const float* W, int K, int N, unsigned short* dst) {
    int total = (K / 32) * (N / 16) * 32;
    mamba_pack_w_bf16<<<dim3((total + 255) / 256), dim3(256), 0, stream>>>(W, K, N, dst);
  };
  auto packA = [&](const float* A, int lda, int K) {
    int total = (M_ROWS / 16) * (K / 32) * 32;
    mamba_pack_a_bf16<<<dim3((total + 255) / 256), dim3(256), 0, stream>>>(A, lda, K, pA, total);
  };

  packW(W_in,  512, 1024, pWin);
  packW(W_xp,  512,   64, pWxp);
  packW(W_dt,   32,  512, pWdt);
  packW(W_out, 1024, 512, pWout);

  // K1: xz = hs @ W_in            (32768 x 512 x 1024)
  packA(hs, D_MODEL, 512);
  mamba_gemm_wmma_bf16<0><<<dim3(1024 / 64, M_ROWS / 128), dim3(256), 0, stream>>>(
      pA, pWin, bufA, D_INNER, 512, 1024, nullptr);

  // K2: depthwise conv + silu (x -> xcv, z -> bufB[:,512:])
  {
    size_t total = (size_t)M_ROWS * D_HALF;
    mamba_dwconv_silu<<<dim3((unsigned)((total + 255) / 256)), dim3(256), 0, stream>>>(
        bufA, K_x, K_z, xcv, bufB);
  }

  // K3: x_dbl = xcv @ W_xproj     (32768 x 512 x 64)
  packA(xcv, D_HALF, 512);
  mamba_gemm_wmma_bf16<0><<<dim3(64 / 64, M_ROWS / 128), dim3(256), 0, stream>>>(
      pA, pWxp, xdbl, 64, 512, 64, nullptr);

  // K4: delta = softplus(x_dbl[:, :32] @ W_dt + b_dt)   (32768 x 32 x 512)
  packA(xdbl, 64, 32);
  mamba_gemm_wmma_bf16<1><<<dim3(512 / 64, M_ROWS / 128), dim3(256), 0, stream>>>(
      pA, pWdt, deltab, D_HALF, 32, 512, b_dt);

  // K5: selective scan -> y into bufB[:, :512]
  mamba_ssm_scan<<<dim3(B_SZ), dim3(D_HALF), 0, stream>>>(
      deltab, xcv, xdbl, A_log, Dvec, bufB);

  // K6: out = [y | z] @ W_out     (32768 x 1024 x 512)
  packA(bufB, D_INNER, 1024);
  mamba_gemm_wmma_bf16<0><<<dim3(512 / 64, M_ROWS / 128), dim3(256), 0, stream>>>(
      pA, pWout, out, D_MODEL, 1024, 512, nullptr);
}
